// MixBlock_3607772529146
// MI455X (gfx1250) — matrix operand order repaired and checked by reading, not compiled, not run on hardware
//
#include <hip/hip_runtime.h>
#include <hip/hip_bf16.h>

// ---------------------------------------------------------------------------
// MixBlock (dual cross-attention) for MI455X / gfx1250, wave32 + WMMA bf16.
//
// Shapes: B=16, C=728, W=H=38, HW=1444.
// Pipeline:
//   1) conv_w   : pack [Wq;Wk] -> bf16 [MP=1472][KP=736], zero padded + bias vec
//   2) conv_xT  : x f32 [B,728,1444] -> bf16 xbT [B][NP=1472][KP=736]
//                 (transposed via LDS tile so GEMM B-staging is all b128)
//   3) gemm_qk  : qk[b] = Wbf x xb[b]^T-packed, double-buffered LDS
//                 (async global->LDS when available), WMMA bf16 f32-accum,
//                 coalesced b128 epilogue through an aliased LDS tile
//   4) attn_fuse: per (b,c): energy = qF.kF^T + qL.kL^T (WMMA, A.B^T == row
//                 reads for both operands), softmax(38), fused dual output
//                 with folded dw+BN affine and 2*sigmoid(gamma)-1 gates.
// ---------------------------------------------------------------------------

#define BB 16
#define CC 728
#define WD 38
#define HW 1444            // 38*38
#define MP 1472            // 2*C padded to 23*64
#define KP 736             // C padded to 23*32
#define NP 1472            // HW padded to 23*64
#define KTILES 23          // KP/32
#define BN_EPS 1e-5f

// CDNA5 async global->LDS staging (GLOBAL_LOAD_ASYNC_TO_LDS_B128, ASYNCcnt).
// Guarded: falls back to regs+ds_store_b128 double buffering if the builtins
// are not declared by this toolchain.
#if defined(__has_builtin)
#  if __has_builtin(__builtin_amdgcn_global_load_async_to_lds_b128) && \
      __has_builtin(__builtin_amdgcn_s_wait_asynccnt)
#    define MIX_ASYNC_LDS 1
#  endif
#endif
#ifndef MIX_ASYNC_LDS
#  define MIX_ASYNC_LDS 0
#endif

typedef __bf16          v16bf  __attribute__((ext_vector_type(16)));
typedef float           v8f    __attribute__((ext_vector_type(8)));
typedef unsigned int    u32x4  __attribute__((ext_vector_type(4)));

__device__ __forceinline__ unsigned short f2bf(float f) {
  unsigned u = __builtin_bit_cast(unsigned, f);
  u += 0x7FFFu + ((u >> 16) & 1u);          // round-to-nearest-even
  return (unsigned short)(u >> 16);
}

union FragU { u32x4 q[2]; v16bf v; };

// A fragment (16x32 bf16, MxK): lane m=l&15; halves e0..7 -> K=hi*8..+7,
// e8..15 -> K=hi*8+16..+23  (ISA 7.12.2 "16-bit A-Matrix 16x32").
__device__ __forceinline__ v16bf load_frag_a(const unsigned short* row, int hi) {
  FragU u;
  u.q[0] = *(const u32x4*)(row + hi * 8);
  u.q[1] = *(const u32x4*)(row + hi * 8 + 16);
  return u.v;
}
// B fragment (32x16 bf16, KxN): lane n=l&15; same K order as A:
// e0..7 -> K=hi*8..+7, e8..15 -> K=hi*8+16..+23.
__device__ __forceinline__ v16bf load_frag_b(const unsigned short* row, int hi) {
  FragU u;
  u.q[0] = *(const u32x4*)(row + hi * 8);
  u.q[1] = *(const u32x4*)(row + hi * 8 + 16);
  return u.v;
}

__device__ __forceinline__ v8f zero8() {
  v8f z = {0.f, 0.f, 0.f, 0.f, 0.f, 0.f, 0.f, 0.f};
  return z;
}

#if MIX_ASYNC_LDS
// Builtin signature (from compiler diagnostic): param0 is a non-const
// pointer to GCC-style int 4-vector in AS1 (printed as "__device__");
// param1 symmetric in AS3; then imm offset, imm cpol.
typedef int v4i_g __attribute__((vector_size(16)));
typedef __attribute__((address_space(1))) v4i_g as1_v4i;
typedef __attribute__((address_space(3))) v4i_g as3_v4i;

__device__ __forceinline__ void async_g2l_b128(const unsigned short* g,
                                               unsigned short* l) {
  __builtin_amdgcn_global_load_async_to_lds_b128(
      (as1_v4i*)(void*)g, (as3_v4i*)(void*)l, 0, 0);
}
#endif

// ------------------------- weight / input packing --------------------------

__global__ __launch_bounds__(256) void conv_w(
    const float* __restrict__ Wq, const float* __restrict__ Wk,
    const float* __restrict__ bq, const float* __restrict__ bk,
    unsigned short* __restrict__ Wbf, float* __restrict__ biasp) {
  int idx = blockIdx.x * 256 + threadIdx.x;
  if (idx >= MP * KP) return;
  int m = idx / KP, k = idx - m * KP;
  float v = 0.f;
  if (k < CC) {
    if (m < CC)            v = Wq[m * CC + k];
    else if (m < 2 * CC)   v = Wk[(m - CC) * CC + k];
  }
  Wbf[idx] = f2bf(v);
  if (k == 0) {
    float bv = 0.f;
    if (m < CC) bv = bq[m]; else if (m < 2 * CC) bv = bk[m - CC];
    biasp[m] = bv;
  }
}

// Tiled transpose + bf16 pack: x [B][C][HW] f32 -> xbT [B][NP][KP] bf16.
// 32x32 tile through LDS; reads coalesced along n, writes coalesced along k.
__global__ __launch_bounds__(256) void conv_xT(
    const float* __restrict__ x, unsigned short* __restrict__ xbT) {
  __shared__ float tile[32][33];
  const int b  = blockIdx.z;
  const int k0 = blockIdx.y * 32;   // channel (GEMM K)
  const int n0 = blockIdx.x * 32;   // spatial (GEMM N)
  const int tid = threadIdx.x;
  const int r  = tid >> 5;          // 0..7
  const int cc = tid & 31;
#pragma unroll
  for (int p = 0; p < 4; ++p) {
    int k = k0 + r + p * 8;
    int n = n0 + cc;
    float v = 0.f;
    if (k < CC && n < HW) v = x[((size_t)b * CC + k) * HW + n];
    tile[r + p * 8][cc] = v;
  }
  __syncthreads();
#pragma unroll
  for (int p = 0; p < 4; ++p) {
    int nr = r + p * 8;
    xbT[((size_t)b * NP + n0 + nr) * KP + k0 + cc] = f2bf(tile[cc][nr]);
  }
}

// ------------------------------ conv1x1 GEMM -------------------------------
// qk[b] (MPxNP bf16) = Wbf (MPxKP) * X (KPxNP, packed as xbT[N][K]) + bias.
// Block tile 64x64, 4 waves, double-buffered LDS, all-b128 staging.

__global__ __launch_bounds__(128) void gemm_qk(
    const unsigned short* __restrict__ Wbf,
    const unsigned short* __restrict__ xbT,   // [B][NP][KP]
    const float* __restrict__ biasp,
    unsigned short* __restrict__ qk) {
  // k-loop double buffers alias the epilogue output tile (dead afterwards)
  __shared__ __align__(16) union SmemU {
    struct { unsigned short A[2][64][32]; unsigned short B[2][64][32]; } t;
    unsigned short C[64][72];   // 72-pad: rows stay 16B aligned for b128
  } sm;

  const int b  = blockIdx.z;
  const int m0 = blockIdx.y * 64;
  const int n0 = blockIdx.x * 64;
  const unsigned short* __restrict__ Bsrc = xbT + (size_t)b * NP * KP;
  unsigned short* __restrict__ Cdst = qk + (size_t)b * MP * NP;

  const int tid  = threadIdx.x;
  const int wid  = tid >> 5;
  const int lane = tid & 31;
  const int l15  = lane & 15;
  const int hi   = lane >> 4;
  const int wy   = wid >> 1;
  const int wx   = wid & 1;

  // staging coords: row tid>>1, two contiguous 16B chunks at (tid&1)*16
  const int sr = tid >> 1;
  const int sc = (tid & 1) * 16;
  const unsigned short* gA = Wbf + (size_t)(m0 + sr) * KP + sc;
  const unsigned short* gB = Bsrc + (size_t)(n0 + sr) * KP + sc;

  v8f acc[2][2];
  acc[0][0] = zero8(); acc[0][1] = zero8();
  acc[1][0] = zero8(); acc[1][1] = zero8();

#if MIX_ASYNC_LDS
  // prologue: DMA tile 0 straight into buffer 0
  async_g2l_b128(gA,     &sm.t.A[0][sr][sc]);
  async_g2l_b128(gA + 8, &sm.t.A[0][sr][sc + 8]);
  async_g2l_b128(gB,     &sm.t.B[0][sr][sc]);
  async_g2l_b128(gB + 8, &sm.t.B[0][sr][sc + 8]);
  __builtin_amdgcn_s_wait_asynccnt(0);
  __syncthreads();

  for (int kt = 0; kt < KTILES; ++kt) {
    const int cur = kt & 1;
    if (kt + 1 < KTILES) {            // DMA next tile while computing
      size_t ko = (size_t)(kt + 1) * 32;
      async_g2l_b128(gA + ko,     &sm.t.A[1 - cur][sr][sc]);
      async_g2l_b128(gA + ko + 8, &sm.t.A[1 - cur][sr][sc + 8]);
      async_g2l_b128(gB + ko,     &sm.t.B[1 - cur][sr][sc]);
      async_g2l_b128(gB + ko + 8, &sm.t.B[1 - cur][sr][sc + 8]);
    }

    v16bf afr[2], bfr[2];
#pragma unroll
    for (int mt = 0; mt < 2; ++mt)
      afr[mt] = load_frag_a(&sm.t.A[cur][wy * 32 + mt * 16 + l15][0], hi);
#pragma unroll
    for (int nt = 0; nt < 2; ++nt)
      bfr[nt] = load_frag_b(&sm.t.B[cur][wx * 32 + nt * 16 + l15][0], hi);

#pragma unroll
    for (int mt = 0; mt < 2; ++mt)
#pragma unroll
      for (int nt = 0; nt < 2; ++nt)
        acc[mt][nt] = __builtin_amdgcn_wmma_f32_16x16x32_bf16(
            false, afr[mt], false, bfr[nt], (short)0, acc[mt][nt], false, false);

    __builtin_amdgcn_s_wait_asynccnt(0);   // after compute: DMA overlapped
    __syncthreads();
  }
#else
  u32x4 ra0, ra1, rb0, rb1;
  // prologue: tile 0 -> regs -> buffer 0
  ra0 = *(const u32x4*)(gA);
  ra1 = *(const u32x4*)(gA + 8);
  rb0 = *(const u32x4*)(gB);
  rb1 = *(const u32x4*)(gB + 8);
  *(u32x4*)(&sm.t.A[0][sr][sc])     = ra0;
  *(u32x4*)(&sm.t.A[0][sr][sc + 8]) = ra1;
  *(u32x4*)(&sm.t.B[0][sr][sc])     = rb0;
  *(u32x4*)(&sm.t.B[0][sr][sc + 8]) = rb1;
  __syncthreads();

  for (int kt = 0; kt < KTILES; ++kt) {
    const int cur = kt & 1;
    const bool more = (kt + 1 < KTILES);
    if (more) {                       // fetch next tile while computing
      size_t ko = (size_t)(kt + 1) * 32;
      ra0 = *(const u32x4*)(gA + ko);
      ra1 = *(const u32x4*)(gA + ko + 8);
      rb0 = *(const u32x4*)(gB + ko);
      rb1 = *(const u32x4*)(gB + ko + 8);
    }

    v16bf afr[2], bfr[2];
#pragma unroll
    for (int mt = 0; mt < 2; ++mt)
      afr[mt] = load_frag_a(&sm.t.A[cur][wy * 32 + mt * 16 + l15][0], hi);
#pragma unroll
    for (int nt = 0; nt < 2; ++nt)
      bfr[nt] = load_frag_b(&sm.t.B[cur][wx * 32 + nt * 16 + l15][0], hi);

#pragma unroll
    for (int mt = 0; mt < 2; ++mt)
#pragma unroll
      for (int nt = 0; nt < 2; ++nt)
        acc[mt][nt] = __builtin_amdgcn_wmma_f32_16x16x32_bf16(
            false, afr[mt], false, bfr[nt], (short)0, acc[mt][nt], false, false);

    if (more) {
      *(u32x4*)(&sm.t.A[1 - cur][sr][sc])     = ra0;
      *(u32x4*)(&sm.t.A[1 - cur][sr][sc + 8]) = ra1;
      *(u32x4*)(&sm.t.B[1 - cur][sr][sc])     = rb0;
      *(u32x4*)(&sm.t.B[1 - cur][sr][sc + 8]) = rb1;
    }
    __syncthreads();
  }
#endif

  // epilogue: acc + bias -> bf16 LDS tile (aliases dead k-loop buffers),
  // then fully coalesced b128 row stores (each wave writes 4 whole rows).
#pragma unroll
  for (int mt = 0; mt < 2; ++mt)
#pragma unroll
    for (int nt = 0; nt < 2; ++nt) {
      int ncol = wx * 32 + nt * 16 + l15;
#pragma unroll
      for (int r = 0; r < 8; ++r) {
        int mrow = wy * 32 + mt * 16 + hi * 8 + r;
        sm.C[mrow][ncol] = f2bf(acc[mt][nt][r] + biasp[m0 + mrow]);
      }
    }
  __syncthreads();
#pragma unroll
  for (int ch = tid; ch < 512; ch += 128) {
    int row = ch >> 3;                 // 8 x 16B chunks per 64-col row
    int cb  = (ch & 7) * 8;
    u32x4 v = *(const u32x4*)(&sm.C[row][cb]);
    *(u32x4*)(Cdst + (size_t)(m0 + row) * NP + (n0 + cb)) = v;
  }
}

// ----------------- energy + softmax + fused dual output --------------------
// One block per (b,c). energy = qF.kF^T + qL.kL^T: since both stored row
// major, both WMMA fragments are contiguous row reads (A.B^T duality).

__global__ __launch_bounds__(128) void attn_fuse(
    const unsigned short* __restrict__ qkF, const unsigned short* __restrict__ qkL,
    const float* __restrict__ xF, const float* __restrict__ xL,
    const float* __restrict__ gammaF, const float* __restrict__ gammaL,
    const float* __restrict__ dwFw, const float* __restrict__ dwFb,
    const float* __restrict__ bnFs, const float* __restrict__ bnFb,
    const float* __restrict__ bnFm, const float* __restrict__ bnFv,
    const float* __restrict__ dwLw, const float* __restrict__ dwLb,
    const float* __restrict__ bnLs, const float* __restrict__ bnLb,
    const float* __restrict__ bnLm, const float* __restrict__ bnLv,
    float* __restrict__ yF, float* __restrict__ yL) {
  __shared__ __align__(16) unsigned short sQ[2][48][64];  // [fad/lfs][i][j]
  __shared__ __align__(16) unsigned short sK[2][48][64];
  __shared__ float sE[48][49];

  const int bc = blockIdx.x;
  const int b  = bc / CC;
  const int c  = bc - b * CC;

  const int tid  = threadIdx.x;
  const int wid  = tid >> 5;
  const int lane = tid & 31;
  const int l15  = lane & 15;
  const int hi   = lane >> 4;

  // folded dw+BN affine and gamma gates (all-lane redundant scalar math)
  const float invF = bnFs[c] * rsqrtf(bnFv[c] + BN_EPS);
  const float A_fad = dwFw[c] * invF;
  const float D_fad = (dwFb[c] - bnFm[c]) * invF + bnFb[c];
  const float invL = bnLs[c] * rsqrtf(bnLv[c] + BN_EPS);
  const float A_lfs = dwLw[c] * invL;
  const float D_lfs = (dwLb[c] - bnLm[c]) * invL + bnLb[c];
  const float g_fad = 2.f / (1.f + __expf(-gammaF[0])) - 1.f;
  const float g_lfs = 2.f / (1.f + __expf(-gammaL[0])) - 1.f;

  // zero padded LDS tiles, then fill 38x38 payloads
  {
    unsigned* zq = (unsigned*)sQ;
    unsigned* zk = (unsigned*)sK;
    for (int i = tid; i < 2 * 48 * 64 / 2; i += 128) { zq[i] = 0u; zk[i] = 0u; }
  }
  __syncthreads();
  {
    const unsigned short* qF = qkF + ((size_t)b * MP + c) * NP;
    const unsigned short* kF = qkF + ((size_t)b * MP + CC + c) * NP;
    const unsigned short* qL = qkL + ((size_t)b * MP + c) * NP;
    const unsigned short* kL = qkL + ((size_t)b * MP + CC + c) * NP;
    for (int n = tid; n < HW; n += 128) {
      int i = n / WD, j = n - i * WD;
      sQ[0][i][j] = qF[n];
      sK[0][i][j] = kF[n];
      sQ[1][i][j] = qL[n];
      sK[1][i][j] = kL[n];
    }
  }
  __syncthreads();

  // 9 output tiles (48x48) split across 4 waves; wave-uniform loop so EXEC
  // stays all-ones around WMMA.
  for (int t = wid; t < 9; t += 4) {
    int tm = t / 3, tn = t - tm * 3;
    v8f acc = zero8();
#pragma unroll
    for (int s = 0; s < 2; ++s)
#pragma unroll
      for (int ks = 0; ks < 2; ++ks) {
        v16bf a  = load_frag_a(&sQ[s][tm * 16 + l15][ks * 32], hi);
        v16bf bm = load_frag_b(&sK[s][tn * 16 + l15][ks * 32], hi);
        acc = __builtin_amdgcn_wmma_f32_16x16x32_bf16(
            false, a, false, bm, (short)0, acc, false, false);
      }
    int ncol = tn * 16 + l15;
#pragma unroll
    for (int r = 0; r < 8; ++r)
      sE[tm * 16 + hi * 8 + r][ncol] = acc[r];
  }
  __syncthreads();

  // softmax over 38 columns, rows 0..37 (one row per thread)
  if (tid < WD) {
    float m = -1e30f;
    for (int k = 0; k < WD; ++k) m = fmaxf(m, sE[tid][k]);
    float ssum = 0.f;
    for (int k = 0; k < WD; ++k) {
      float e = __expf(sE[tid][k] - m);
      sE[tid][k] = e;
      ssum += e;
    }
    float inv = 1.f / ssum;
    for (int k = 0; k < WD; ++k) sE[tid][k] *= inv;
  }
  __syncthreads();

  // fused dual output with residual + folded affine
  const size_t base = ((size_t)b * CC + c) * HW;
  for (int n = tid; n < HW; n += 128) {
    int i = n / WD, j = n - i * WD;
    float a  = sE[i][j];
    float vF = xF[base + n];
    float vL = xL[base + n];
    yF[base + n] = vF + A_fad * (vL * a * g_lfs) + D_fad;
    yL[base + n] = vL + A_lfs * (vF * a * g_fad) + D_lfs;
  }
}

// ------------------------------- launcher ----------------------------------

extern "C" void kernel_launch(void* const* d_in, const int* in_sizes, int n_in,
                              void* d_out, int out_size, void* d_ws, size_t ws_size,
                              hipStream_t stream) {
  const float* x_FAD   = (const float*)d_in[0];
  const float* x_LFS   = (const float*)d_in[1];
  const float* Wq_fad  = (const float*)d_in[2];
  const float* bq_fad  = (const float*)d_in[3];
  const float* Wq_lfs  = (const float*)d_in[4];
  const float* bq_lfs  = (const float*)d_in[5];
  const float* Wk_fad  = (const float*)d_in[6];
  const float* bk_fad  = (const float*)d_in[7];
  const float* Wk_lfs  = (const float*)d_in[8];
  const float* bk_lfs  = (const float*)d_in[9];
  const float* gamma_fad = (const float*)d_in[10];
  const float* gamma_lfs = (const float*)d_in[11];
  const float* dw_fad_w = (const float*)d_in[12];
  const float* dw_fad_b = (const float*)d_in[13];
  const float* dw_lfs_w = (const float*)d_in[14];
  const float* dw_lfs_b = (const float*)d_in[15];
  const float* fad_bn_s = (const float*)d_in[16];
  const float* fad_bn_b = (const float*)d_in[17];
  const float* fad_bn_m = (const float*)d_in[18];
  const float* fad_bn_v = (const float*)d_in[19];
  const float* lfs_bn_s = (const float*)d_in[20];
  const float* lfs_bn_b = (const float*)d_in[21];
  const float* lfs_bn_m = (const float*)d_in[22];
  const float* lfs_bn_v = (const float*)d_in[23];

  float* yF = (float*)d_out;
  float* yL = yF + (size_t)BB * CC * HW;

  // workspace carve-up (all sizes 256B multiples)
  char* ws = (char*)d_ws;
  size_t off = 0;
  auto carve = [&](size_t bytes) {
    void* p = ws + off;
    off += (bytes + 255) & ~(size_t)255;
    return p;
  };
  const size_t szWbf  = (size_t)MP * KP * sizeof(unsigned short);
  const size_t szBias = (size_t)MP * sizeof(float);
  const size_t szXb   = (size_t)BB * NP * KP * sizeof(unsigned short);
  const size_t szQk   = (size_t)BB * MP * NP * sizeof(unsigned short);

  unsigned short* WbfF = (unsigned short*)carve(szWbf);
  unsigned short* WbfL = (unsigned short*)carve(szWbf);
  float*          biasF = (float*)carve(szBias);
  float*          biasL = (float*)carve(szBias);
  unsigned short* xbF  = (unsigned short*)carve(szXb);
  unsigned short* xbL  = (unsigned short*)carve(szXb);
  unsigned short* qkF  = (unsigned short*)carve(szQk);
  unsigned short* qkL  = (unsigned short*)carve(szQk);

  // 1) pack weights (+bias) to bf16
  {
    int n = MP * KP;
    dim3 g((n + 255) / 256);
    conv_w<<<g, 256, 0, stream>>>(Wq_fad, Wk_fad, bq_fad, bk_fad, WbfF, biasF);
    conv_w<<<g, 256, 0, stream>>>(Wq_lfs, Wk_lfs, bq_lfs, bk_lfs, WbfL, biasL);
  }
  // 2) pack + transpose activations to bf16 [B][N][K]
  {
    dim3 g(NP / 32, KP / 32, BB);
    conv_xT<<<g, 256, 0, stream>>>(x_FAD, xbF);
    conv_xT<<<g, 256, 0, stream>>>(x_LFS, xbL);
  }
  // 3) conv1x1 GEMMs via WMMA bf16 (double-buffered LDS, async DMA staging)
  {
    dim3 g(NP / 64, MP / 64, BB);
    gemm_qk<<<g, 128, 0, stream>>>(WbfF, xbF, biasF, qkF);
    gemm_qk<<<g, 128, 0, stream>>>(WbfL, xbL, biasL, qkL);
  }
  // 4) energy + softmax + fused outputs
  attn_fuse<<<BB * CC, 128, 0, stream>>>(
      qkF, qkL, x_FAD, x_LFS, gamma_fad, gamma_lfs,
      dw_fad_w, dw_fad_b, fad_bn_s, fad_bn_b, fad_bn_m, fad_bn_v,
      dw_lfs_w, dw_lfs_b, lfs_bn_s, lfs_bn_b, lfs_bn_m, lfs_bn_v,
      yF, yL);
}
